// DualAdaptiveQuantizerV7_89781996356273
// MI455X (gfx1250) — matrix operand order repaired and checked by reading, not compiled, not run on hardware
//
#include <hip/hip_runtime.h>
#include <math.h>

typedef __attribute__((ext_vector_type(16))) _Float16 v16h;
typedef __attribute__((ext_vector_type(8)))  float    v8f;

__device__ __forceinline__ v8f wmma16(v16h a, v16h b, v8f c) {
  // D = A(16x32 f16) * B(32x16 f16) + C(16x16 f32)
  return __builtin_amdgcn_wmma_f32_16x16x32_f16(
      /*neg_a=*/false, a, /*neg_b=*/false, b,
      /*c_mod=*/(short)0, c, /*reuse_a=*/false, /*reuse_b=*/false);
}

__constant__ float cBITS[6] = {2.f, 4.f, 6.f, 8.f, 12.f, 16.f};
__constant__ float cQN[6]   = {-2.f, -8.f, -32.f, -128.f, -2048.f, -32768.f};
__constant__ float cQP[6]   = { 1.f,  7.f,  31.f,  127.f,  2047.f,  32767.f};

// One wave processes a tile of 16 links. Block = 128 threads = 4 waves.
__global__ void __launch_bounds__(128, 1)
daq_fused(const float* __restrict__ V,   const float* __restrict__ SIDE,
          const float* __restrict__ SNR, const float* __restrict__ GD,
          const float* __restrict__ GS,
          const float* __restrict__ W1,  const float* __restrict__ B1,
          const float* __restrict__ W2,  const float* __restrict__ B2,
          const float* __restrict__ WD,  const float* __restrict__ BD,
          const float* __restrict__ WS,  const float* __restrict__ BS,
          const float* __restrict__ SDm, const float* __restrict__ SSc,
          float* __restrict__ o_vq, float* __restrict__ o_sq,
          float* __restrict__ o_bits, float* __restrict__ o_wd,
          float* __restrict__ o_ws, float* __restrict__ o_db,
          float* __restrict__ o_sb, int n_tiles)
{
  // per-wave: 16x64 f32 h-staging (1024) + 16x16 logits_d (256) + 16x16 logits_s (256)
  __shared__ float lds[4 * 1536];
  const int  wave = threadIdx.x >> 5;
  const int  lane = threadIdx.x & 31;
  const int  nn   = lane & 15;          // column / row-within-half
  const bool hiL  = lane >= 16;
  float* hbuf = &lds[wave * 1536];
  float* logd = hbuf + 1024;
  float* logs = logd + 256;

  // ---------- build weight B-fragments once per wave ----------
  // B (KxN, 16-bit): lanes 0-15 hold K=0..7,16..23, lanes 16-31 hold K=8..15,24..31; N = lane&15
  v16h w1f[3], w2f[2][3], wdf[2], wsf[2];
#pragma unroll
  for (int t = 0; t < 3; ++t) {
    v16h f{};
#pragma unroll
    for (int e = 0; e < 16; ++e) {
      int k = (hiL ? 8 : 0) + ((e < 8) ? e : e + 8); // K in [0,32)
      f[e] = (_Float16)((k < 7) ? W1[k * 48 + t * 16 + nn] : 0.f);
    }
    w1f[t] = f;
  }
#pragma unroll
  for (int ks = 0; ks < 2; ++ks) {
#pragma unroll
    for (int t = 0; t < 3; ++t) {
      v16h f{};
#pragma unroll
      for (int e = 0; e < 16; ++e) {
        int k = ks * 32 + (hiL ? 8 : 0) + ((e < 8) ? e : e + 8); // K in [0,64) pad
        f[e] = (_Float16)((k < 48) ? W2[k * 48 + t * 16 + nn] : 0.f);
      }
      w2f[ks][t] = f;
    }
    v16h fd{}, fs{};
#pragma unroll
    for (int e = 0; e < 16; ++e) {
      int k = ks * 32 + (hiL ? 8 : 0) + ((e < 8) ? e : e + 8);
      bool ok = (k < 48) && (nn < 6);
      fd[e] = (_Float16)(ok ? WD[k * 6 + nn] : 0.f);
      fs[e] = (_Float16)(ok ? WS[k * 6 + nn] : 0.f);
    }
    wdf[ks] = fd;
    wsf[ks] = fs;
  }

  // bias values (column-indexed; invariant across tiles)
  float b1v[3], b2v[3];
#pragma unroll
  for (int t = 0; t < 3; ++t) { b1v[t] = B1[t * 16 + nn]; b2v[t] = B2[t * 16 + nn]; }
  const float bdv = (nn < 6) ? BD[nn] : 0.f;
  const float bsv = (nn < 6) ? BS[nn] : 0.f;

  // zero the padded h staging buffer once: cols 48..63 stay zero forever,
  // so K-padding of the 48-wide activations is free.
  for (int i = lane; i < 1024; i += 32) hbuf[i] = 0.f;
  asm volatile("s_wait_dscnt 0" ::: "memory");

  const int wgid    = blockIdx.x * 4 + wave;
  const int wstride = gridDim.x * 4;

  for (int tile = wgid; tile < n_tiles; tile += wstride) {
    const int base = tile * 16;

    // ---------- load inputs, build A1 (16x32 f16; only K<7 nonzero) ----------
    float vx = 0.f, vy = 0.f, s0 = 0.f, s1 = 0.f, s2 = 0.f;
    v16h a1{};                                    // lanes 16-31: all K>=8 -> zero
    if (!hiL) {
      const int link = base + nn;
      vx = V[link * 2 + 0];
      vy = V[link * 2 + 1];
      s0 = SIDE[link * 3 + 0];
      s1 = SIDE[link * 3 + 1];
      s2 = SIDE[link * 3 + 2];
      const float sn  = SNR[link];
      const float mag = sqrtf(vx * vx + vy * vy + 1e-10f);
      // pin = [v0, v1, snr, side2, mag, side0, side1]
      a1[0] = (_Float16)vx; a1[1] = (_Float16)vy; a1[2] = (_Float16)sn;
      a1[3] = (_Float16)s2; a1[4] = (_Float16)mag;
      a1[5] = (_Float16)s0; a1[6] = (_Float16)s1;
    }

    // ---------- layer 1: h = relu(pin @ W1 + b1), store to LDS ----------
#pragma unroll
    for (int t = 0; t < 3; ++t) {
      const float bb = b1v[t];
      v8f c = {bb, bb, bb, bb, bb, bb, bb, bb};
      c = wmma16(a1, w1f[t], c);
#pragma unroll
      for (int j = 0; j < 8; ++j)
        hbuf[(j + (hiL ? 8 : 0)) * 64 + t * 16 + nn] = fmaxf(c[j], 0.f);
    }
    asm volatile("s_wait_dscnt 0" ::: "memory");

    // A-fragment (16-bit, 16x32): lane row = nn; e<8 -> K=base8+e, e>=8 -> K=base8+16+(e-8)
    v16h a2[2];
#pragma unroll
    for (int ks = 0; ks < 2; ++ks) {
      v16h f{};
#pragma unroll
      for (int e = 0; e < 16; ++e) {
        const int k = ks * 32 + (hiL ? 8 : 0) + ((e < 8) ? e : e + 8);
        f[e] = (_Float16)hbuf[nn * 64 + k];
      }
      a2[ks] = f;
    }

    // ---------- layer 2: h2 = relu(h @ W2 + b2), store to LDS ----------
#pragma unroll
    for (int t = 0; t < 3; ++t) {
      const float bb = b2v[t];
      v8f c = {bb, bb, bb, bb, bb, bb, bb, bb};
      c = wmma16(a2[0], w2f[0][t], c);
      c = wmma16(a2[1], w2f[1][t], c);
#pragma unroll
      for (int j = 0; j < 8; ++j)
        hbuf[(j + (hiL ? 8 : 0)) * 64 + t * 16 + nn] = fmaxf(c[j], 0.f);
    }
    asm volatile("s_wait_dscnt 0" ::: "memory");

    v16h a3[2];
#pragma unroll
    for (int ks = 0; ks < 2; ++ks) {
      v16h f{};
#pragma unroll
      for (int e = 0; e < 16; ++e) {
        const int k = ks * 32 + (hiL ? 8 : 0) + ((e < 8) ? e : e + 8);
        f[e] = (_Float16)hbuf[nn * 64 + k];
      }
      a3[ks] = f;
    }

    // ---------- heads: logits_d / logits_s (N=6 padded to 16) ----------
    v8f cd = {bdv, bdv, bdv, bdv, bdv, bdv, bdv, bdv};
    cd = wmma16(a3[0], wdf[0], cd);
    cd = wmma16(a3[1], wdf[1], cd);
    v8f cs = {bsv, bsv, bsv, bsv, bsv, bsv, bsv, bsv};
    cs = wmma16(a3[0], wsf[0], cs);
    cs = wmma16(a3[1], wsf[1], cs);
#pragma unroll
    for (int j = 0; j < 8; ++j) {
      const int m = j + (hiL ? 8 : 0);
      logd[m * 16 + nn] = cd[j];
      logs[m * 16 + nn] = cs[j];
    }
    asm volatile("s_wait_dscnt 0" ::: "memory");

    // ---------- per-link finalize (Gumbel argmax + LSQ + outputs) ----------
    if (!hiL) {
      const int link = base + nn;
      int id = 0;  float bestd = -3.4e38f;
      int is = 0;  float bests = -3.4e38f;
#pragma unroll
      for (int i = 0; i < 6; ++i) {
        const float zd = logd[nn * 16 + i] + GD[link * 6 + i];
        if (zd > bestd) { bestd = zd; id = i; }
        const float zs = logs[nn * 16 + i] + GS[link * 6 + i];
        if (zs > bests) { bests = zs; is = i; }
      }
      const float sd  = SDm[id];
      const float ssx = SSc[is];
      const float qnd = cQN[id], qpd = cQP[id];
      const float qns = cQN[is], qps = cQP[is];

      o_vq[link * 2 + 0] = rintf(fminf(fmaxf(vx / sd, qnd), qpd)) * sd;
      o_vq[link * 2 + 1] = rintf(fminf(fmaxf(vy / sd, qnd), qpd)) * sd;
      o_sq[link * 3 + 0] = rintf(fminf(fmaxf(s0 / ssx, qns), qps)) * ssx;
      o_sq[link * 3 + 1] = rintf(fminf(fmaxf(s1 / ssx, qns), qps)) * ssx;
      o_sq[link * 3 + 2] = rintf(fminf(fmaxf(s2 / ssx, qns), qps)) * ssx;
#pragma unroll
      for (int i = 0; i < 6; ++i) {
        o_wd[link * 6 + i] = (i == id) ? 1.f : 0.f;
        o_ws[link * 6 + i] = (i == is) ? 1.f : 0.f;
      }
      const float db = 2.f * cBITS[id];
      const float sb = 3.f * cBITS[is];
      o_db[link]   = db;
      o_sb[link]   = sb;
      o_bits[link] = db + sb;
    }
  }
}

extern "C" void kernel_launch(void* const* d_in, const int* in_sizes, int n_in,
                              void* d_out, int out_size, void* d_ws, size_t ws_size,
                              hipStream_t stream) {
  const float* V    = (const float*)d_in[0];
  const float* SIDE = (const float*)d_in[1];
  const float* SNR  = (const float*)d_in[2];
  const float* GD   = (const float*)d_in[3];
  const float* GS   = (const float*)d_in[4];
  const float* W1   = (const float*)d_in[5];
  const float* B1   = (const float*)d_in[6];
  const float* W2   = (const float*)d_in[7];
  const float* B2   = (const float*)d_in[8];
  const float* WD   = (const float*)d_in[9];
  const float* BD   = (const float*)d_in[10];
  const float* WS   = (const float*)d_in[11];
  const float* BS   = (const float*)d_in[12];
  const float* SDm  = (const float*)d_in[13];
  const float* SSc  = (const float*)d_in[14];

  const int n_links = in_sizes[0] / 2;      // (B,L,K,2) flat
  const int n_tiles = n_links / 16;         // 16 links per wave-tile

  const size_t NL = (size_t)n_links;
  float* out    = (float*)d_out;
  float* o_vq   = out;                      // 2*NL
  float* o_sq   = o_vq  + NL * 2;           // 3*NL
  float* o_bits = o_sq  + NL * 3;           // NL
  float* o_wd   = o_bits + NL;              // 6*NL
  float* o_ws   = o_wd  + NL * 6;           // 6*NL
  float* o_db   = o_ws  + NL * 6;           // NL
  float* o_sb   = o_db  + NL;               // NL

  int blocks = 2048;                        // 8192 waves -> ~4 tiles/wave
  const int maxb = (n_tiles + 3) / 4;
  if (blocks > maxb) blocks = maxb;
  if (blocks < 1) blocks = 1;

  daq_fused<<<blocks, 128, 0, stream>>>(V, SIDE, SNR, GD, GS,
                                        W1, B1, W2, B2, WD, BD, WS, BS,
                                        SDm, SSc,
                                        o_vq, o_sq, o_bits, o_wd, o_ws,
                                        o_db, o_sb, n_tiles);
}
